// ExplicitOnlyDiffusionModel_4707284156476
// MI455X (gfx1250) — matrix operand order repaired and checked by reading, not compiled, not run on hardware
//
#include <hip/hip_runtime.h>
#include <math.h>

// ---------------------------------------------------------------------------
// Types for CDNA5 WMMA
// ---------------------------------------------------------------------------
typedef __attribute__((ext_vector_type(16))) __bf16 v16bf;
typedef __attribute__((ext_vector_type(8)))  float  v8f;

#define B_  512
#define S_  500
#define DIN_ 14
#define H_  128
#define N_  256
#define SP_ 512            // padded sequence length (multiple of 64)
#define M_  (B_ * S_)      // 256000 rows, divisible by 128

__device__ __forceinline__ __bf16 f2bf(float f) {
    unsigned u = __float_as_uint(f);
    unsigned r = (u + 0x7FFFu + ((u >> 16) & 1u)) >> 16;
    unsigned short s = (unsigned short)r;
    return __builtin_bit_cast(__bf16, s);
}
__device__ __forceinline__ float bf2f(__bf16 b) {
    unsigned short s = __builtin_bit_cast(unsigned short, b);
    return __uint_as_float(((unsigned)s) << 16);
}
__device__ __forceinline__ float gelu_exact(float x) {
    return 0.5f * x * (1.0f + erff(x * 0.70710678118654752f));
}

// ---------------------------------------------------------------------------
// 1) S4D kernel synthesis: K[l][h][t] = 2*Re( sum_n C'_n * exp(dtA_n * t) )
//    C' = (Cre+iCim)*(exp(dtA)-1)/A.  Stored bf16 [L*H][SP_], zero padded.
// ---------------------------------------------------------------------------
__global__ __launch_bounds__(128) void k_s4_kernel(
    const float* __restrict__ logA, const float* __restrict__ Aim,
    const float* __restrict__ Cre,  const float* __restrict__ Cim,
    const float* __restrict__ logdt, __bf16* __restrict__ Kker)
{
    __shared__ float ar[N_], ai[N_], cr[N_], ci[N_];
    const int lh  = blockIdx.x;          // l*128 + h, 512 blocks
    const int tid = threadIdx.x;
    const float dt = __expf(logdt[lh]);
    const size_t base = (size_t)lh * N_;
    for (int n = tid; n < N_; n += 128) {
        float are = -__expf(logA[base + n]);
        float aim = Aim[base + n];
        float dr = dt * are, di = dt * aim;
        float er = __expf(dr);
        float Er = er * __cosf(di) - 1.0f;
        float Ei = er * __sinf(di);
        float crr = Cre[base + n], cii = Cim[base + n];
        float tr = crr * Er - cii * Ei;
        float ti = crr * Ei + cii * Er;
        float inv = 1.0f / (are * are + aim * aim);
        cr[n] = (tr * are + ti * aim) * inv;
        ci[n] = (ti * are - tr * aim) * inv;
        ar[n] = dr;  // dt*Re(A)
        ai[n] = di;  // dt*Im(A)
    }
    __syncthreads();
    for (int t = tid; t < SP_; t += 128) {
        float acc = 0.0f;
        if (t < S_) {
            float tf = (float)t;
            for (int n = 0; n < N_; ++n) {
                float mag = __expf(ar[n] * tf);
                float ph  = ai[n] * tf;
                acc += mag * (cr[n] * __cosf(ph) - ci[n] * __sinf(ph));
            }
            acc *= 2.0f;
        }
        Kker[(size_t)lh * SP_ + t] = f2bf(acc);
    }
}

// ---------------------------------------------------------------------------
// 2) Input embedding: h = (x*mask+observed) @ W_in + b_in + pe*mask
// ---------------------------------------------------------------------------
__global__ __launch_bounds__(128) void k_embed(
    const float* __restrict__ x, const float* __restrict__ mask,
    const float* __restrict__ observed, const float* __restrict__ W_in,
    const float* __restrict__ b_in, float* __restrict__ h)
{
    __shared__ float xl[DIN_];
    const int bs = blockIdx.x;            // b*S + s
    const int tid = threadIdx.x;
    const float m = mask[bs];
    if (tid < DIN_) xl[tid] = x[(size_t)bs * DIN_ + tid] * m + observed[(size_t)bs * DIN_ + tid];
    __syncthreads();
    float acc = b_in[tid];
    #pragma unroll
    for (int k = 0; k < DIN_; ++k) acc += xl[k] * W_in[k * H_ + tid];
    const int ss = bs % S_;
    const float j = (float)((tid < 64) ? tid : tid - 64);
    const float dv = __expf(-logf(10000.0f) * 2.0f * j / (float)H_);
    const float ang = (float)ss * dv;
    const float pe = (tid < 64) ? __sinf(ang) : __cosf(ang);
    h[(size_t)bs * H_ + tid] = acc + pe * m;
}

// ---------------------------------------------------------------------------
// 3) LayerNorm over H=128. mode 0: row-major bf16 [M,H]. mode 1: transposed
//    bf16 [B][H][SP_] (contiguous-in-time reads for the Toeplitz conv).
// ---------------------------------------------------------------------------
__global__ __launch_bounds__(128) void k_ln(
    const float* __restrict__ h, const float* __restrict__ g,
    const float* __restrict__ bb, __bf16* __restrict__ out, int mode)
{
    __shared__ float red[128];
    const int bs = blockIdx.x;
    const int tid = threadIdx.x;
    const float v = h[(size_t)bs * H_ + tid];
    red[tid] = v; __syncthreads();
    #pragma unroll
    for (int o = 64; o > 0; o >>= 1) { if (tid < o) red[tid] += red[tid + o]; __syncthreads(); }
    const float mean = red[0] * (1.0f / H_);
    __syncthreads();
    const float d = v - mean;
    red[tid] = d * d; __syncthreads();
    #pragma unroll
    for (int o = 64; o > 0; o >>= 1) { if (tid < o) red[tid] += red[tid + o]; __syncthreads(); }
    const float r = rsqrtf(red[0] * (1.0f / H_) + 1e-5f);
    const float z = d * r * g[tid] + bb[tid];
    if (mode == 0) {
        out[(size_t)bs * H_ + tid] = f2bf(z);
    } else {
        const int b = bs / S_, s = bs % S_;
        out[((size_t)(b * H_ + tid)) * SP_ + s] = f2bf(z);
    }
}

// ---------------------------------------------------------------------------
// 4) Causal Toeplitz conv via WMMA bf16, fused + D*u and GELU.
//    Orientation: M = time (16 rows), N = batch (64 cols), one channel h.
//    A = Toeplitz-of-K fragment, gathered BRANCHLESSLY from a reversed,
//    zero-padded LDS copy R where R[511-t] = K[t], R[512..1023] = 0, so
//    A[m=t, kappa] = K[t - tau] = R[511 - t + tau]  (unconditional read,
//    contiguous in kappa).  A is shared by all 4 batch sub-tiles.
//    B = z fragment: half-slots map to K=kofs+i -> contiguous tau, so each
//    B fragment is two aligned 16-byte global loads from zt [B][H][SP_].
// ---------------------------------------------------------------------------
__global__ __launch_bounds__(32) void k_conv(
    const __bf16* __restrict__ zt, const __bf16* __restrict__ Kk,
    const float* __restrict__ D, __bf16* __restrict__ outA)
{
    __shared__ __bf16 R[1024];
    const int lane = threadIdx.x;
    const int hch  = blockIdx.z;
    const int b0   = blockIdx.x * 64;    // batch tile
    const int t0   = blockIdx.y * 16;    // time tile
    const __bf16* Krow = Kk + (size_t)hch * SP_;
    {
        unsigned short z0 = 0;
        for (int i = lane; i < 1024; i += 32)
            R[i] = (i < SP_) ? Krow[SP_ - 1 - i] : __builtin_bit_cast(__bf16, z0);
    }
    __syncthreads();

    const float Dh = D[hch];
    const int ln15  = lane & 15;
    const int hi    = (lane < 16) ? 0 : 16;   // A-fragment kappa base
    const int kofs  = (lane < 16) ? 0 : 16;   // B-fragment kappa base
    const int trow  = t0 + ln15;              // A row -> absolute time
    // per-lane base into reversed kernel: R[base + kappa] = K[trow - tau0 - kappa]
    const int rbase0 = (SP_ - 1) - trow + hi;

    // B-fragment row pointers (4 batch sub-tiles)
    const __bf16* zr0 = zt + ((size_t)(b0 + 0  + ln15) * H_ + hch) * SP_;
    const __bf16* zr1 = zt + ((size_t)(b0 + 16 + ln15) * H_ + hch) * SP_;
    const __bf16* zr2 = zt + ((size_t)(b0 + 32 + ln15) * H_ + hch) * SP_;
    const __bf16* zr3 = zt + ((size_t)(b0 + 48 + ln15) * H_ + hch) * SP_;

    v8f acc[4] = {};
    for (int tau0 = 0; tau0 <= t0 + 15; tau0 += 32) {
        // --- A fragment: Toeplitz of K, branchless LDS gather ------------
        v16bf afv;
        const int rb = rbase0 + tau0;
        #pragma unroll
        for (int i = 0; i < 16; ++i)
            afv[i] = R[rb + i];
        // --- B fragments: contiguous z reads, two uint4 each -------------
        union { uint4 u[2]; v16bf v; } bf0, bf1, bf2, bf3;
        {
            const uint4* q0 = (const uint4*)(zr0 + tau0 + kofs);
            const uint4* q1 = (const uint4*)(zr1 + tau0 + kofs);
            const uint4* q2 = (const uint4*)(zr2 + tau0 + kofs);
            const uint4* q3 = (const uint4*)(zr3 + tau0 + kofs);
            bf0.u[0] = q0[0]; bf0.u[1] = q0[1];
            bf1.u[0] = q1[0]; bf1.u[1] = q1[1];
            bf2.u[0] = q2[0]; bf2.u[1] = q2[1];
            bf3.u[0] = q3[0]; bf3.u[1] = q3[1];
        }
        acc[0] = __builtin_amdgcn_wmma_f32_16x16x32_bf16(false, afv, false, bf0.v, (short)0, acc[0], false, false);
        acc[1] = __builtin_amdgcn_wmma_f32_16x16x32_bf16(false, afv, false, bf1.v, (short)0, acc[1], false, false);
        acc[2] = __builtin_amdgcn_wmma_f32_16x16x32_bf16(false, afv, false, bf2.v, (short)0, acc[2], false, false);
        acc[3] = __builtin_amdgcn_wmma_f32_16x16x32_bf16(false, afv, false, bf3.v, (short)0, acc[3], false, false);
    }

    // epilogue: y = conv + D*u, GELU, scatter to row-major [B*S][H] bf16
    const int mofs = (lane < 16) ? 0 : 8;
    #pragma unroll
    for (int r = 0; r < 8; ++r) {
        const int tt = t0 + r + mofs;
        if (tt < S_) {
            #pragma unroll
            for (int sub = 0; sub < 4; ++sub) {
                const int bbi = b0 + sub * 16 + ln15;
                const float u = bf2f(zt[((size_t)bbi * H_ + hch) * SP_ + tt]);
                const float y = acc[sub][r] + Dh * u;
                outA[((size_t)bbi * S_ + tt) * H_ + hch] = f2bf(gelu_exact(y));
            }
        }
    }
}

// ---------------------------------------------------------------------------
// 5) Generic WMMA bf16 GEMM:  out = act(A[M,K] @ W[K,Ntot] + bias)
//    Block = 8 waves, 128 rows x 128 cols.  W tile staged TRANSPOSED in LDS
//    (WldsT[n][k], row stride K+8 -> conflict-free), so each B fragment is
//    two aligned ds_load_b128 (half-slots contiguous in k).
//    accout!=null -> f32 accumulate (h+=).  bfout!=null -> bf16 store.
// ---------------------------------------------------------------------------
__global__ __launch_bounds__(256) void k_gemm(
    const __bf16* __restrict__ A, int K, const float* __restrict__ W, int Ntot,
    const float* __restrict__ bias, float* __restrict__ accout,
    __bf16* __restrict__ bfout, int dogelu)
{
    extern __shared__ __bf16 WldsT[];            // [128][K+8]
    const int KP = K + 8;
    const int tid = threadIdx.x;
    const int ncol0 = blockIdx.y * 128;
    for (int idx = tid; idx < K * 128; idx += 256) {
        const int n = idx & 127, k = idx >> 7;   // coalesced global read over n
        WldsT[n * KP + k] = f2bf(W[(size_t)k * Ntot + ncol0 + n]);
    }
    __syncthreads();

    const int lane = tid & 31, w = tid >> 5;
    const size_t m0 = (size_t)blockIdx.x * 128 + (size_t)w * 16;
    const int ln15 = lane & 15;
    const int abase = (lane < 16) ? 0 : 16;
    const int kofs  = (lane < 16) ? 0 : 16;
    const __bf16* arow = A + (m0 + ln15) * K;

    v8f acc[2][4] = {};
    for (int k0 = 0; k0 < K; k0 += 32) {
        union { uint4 u[2]; v16bf v; } af;
        const uint4* p = (const uint4*)(arow + k0 + abase);
        af.u[0] = p[0];
        af.u[1] = p[1];
        #pragma unroll
        for (int cg = 0; cg < 2; ++cg) {
            #pragma unroll
            for (int sub = 0; sub < 4; ++sub) {
                const int n = cg * 64 + sub * 16 + ln15;
                union { uint4 u[2]; v16bf v; } bf;
                const uint4* q = (const uint4*)(&WldsT[n * KP + k0 + kofs]);
                bf.u[0] = q[0];
                bf.u[1] = q[1];
                acc[cg][sub] = __builtin_amdgcn_wmma_f32_16x16x32_bf16(
                    false, af.v, false, bf.v, (short)0, acc[cg][sub], false, false);
            }
        }
    }
    const int mofs = (lane < 16) ? 0 : 8;
    #pragma unroll
    for (int cg = 0; cg < 2; ++cg) {
        #pragma unroll
        for (int sub = 0; sub < 4; ++sub) {
            const int ncol = ncol0 + cg * 64 + sub * 16 + ln15;
            #pragma unroll
            for (int r = 0; r < 8; ++r) {
                const size_t row = m0 + r + mofs;
                float v = acc[cg][sub][r] + bias[ncol];
                if (dogelu) v = gelu_exact(v);
                if (accout) accout[row * 128 + ncol] += v;
                if (bfout)  bfout[row * Ntot + ncol] = f2bf(v);
            }
        }
    }
}

// ---------------------------------------------------------------------------
// 6) Diffusion time embedding MLP (tiny): te[b] = silu(sin/cos @ W1+b1)@W2+b2
// ---------------------------------------------------------------------------
__global__ __launch_bounds__(256) void k_time_mlp(
    const int* __restrict__ timesteps, const float* __restrict__ W1,
    const float* __restrict__ b1, const float* __restrict__ W2,
    const float* __restrict__ b2, float* __restrict__ te)
{
    __shared__ float ti[256], hid[256];
    const int b = blockIdx.x, tid = threadIdx.x;
    const float t = (float)timesteps[b];
    const int j = (tid < 128) ? tid : (tid - 128);
    const float fr = __expf(-logf(10000.0f) / 127.0f * (float)j);
    ti[tid] = (tid < 128) ? __sinf(t * fr) : __cosf(t * fr);
    __syncthreads();
    float a = b1[tid];
    for (int k = 0; k < 256; ++k) a += ti[k] * W1[k * 256 + tid];
    hid[tid] = a / (1.0f + __expf(-a));
    __syncthreads();
    if (tid < 128) {
        float o = b2[tid];
        for (int k = 0; k < 256; ++k) o += hid[k] * W2[k * 128 + tid];
        te[(size_t)b * H_ + tid] = o;
    }
}

__global__ __launch_bounds__(256) void k_add_te(
    float* __restrict__ h, const float* __restrict__ te)
{
    const size_t idx = (size_t)blockIdx.x * 256 + threadIdx.x;
    const int b = (int)(idx / (S_ * H_));
    const int hh = (int)(idx & (H_ - 1));
    h[idx] += te[(size_t)b * H_ + hh];
}

// ---------------------------------------------------------------------------
// 7) Output projection: out[M,14] = h[M,128] @ W_out[128,14] + b_out
// ---------------------------------------------------------------------------
__global__ __launch_bounds__(224) void k_out(
    const float* __restrict__ h, const float* __restrict__ W_out,
    const float* __restrict__ b_out, float* __restrict__ out)
{
    __shared__ float rows[16][H_];
    const int tid = threadIdx.x;
    const size_t r0 = (size_t)blockIdx.x * 16;
    for (int i = tid; i < 16 * H_; i += 224)
        rows[i >> 7][i & 127] = h[r0 * H_ + i];
    __syncthreads();
    const int r = tid / DIN_, d = tid % DIN_;
    float o = b_out[d];
    #pragma unroll 8
    for (int k = 0; k < H_; ++k) o += rows[r][k] * W_out[k * DIN_ + d];
    out[(r0 + r) * DIN_ + d] = o;
}

// ---------------------------------------------------------------------------
// Host-side orchestration
// ---------------------------------------------------------------------------
extern "C" void kernel_launch(void* const* d_in, const int* in_sizes, int n_in,
                              void* d_out, int out_size, void* d_ws, size_t ws_size,
                              hipStream_t stream) {
    const float* x        = (const float*)d_in[0];
    const int*   tsteps   = (const int*)  d_in[1];
    const float* mask     = (const float*)d_in[2];
    const float* observed = (const float*)d_in[3];
    const float* W_in     = (const float*)d_in[4];
    const float* b_in     = (const float*)d_in[5];
    const float* time_W1  = (const float*)d_in[6];
    const float* time_b1  = (const float*)d_in[7];
    const float* time_W2  = (const float*)d_in[8];
    const float* time_b2  = (const float*)d_in[9];
    const float* s4_ln_g  = (const float*)d_in[10];
    const float* s4_ln_b  = (const float*)d_in[11];
    const float* s4_logA  = (const float*)d_in[12];
    const float* s4_Aim   = (const float*)d_in[13];
    const float* s4_Cre   = (const float*)d_in[14];
    const float* s4_Cim   = (const float*)d_in[15];
    const float* s4_D     = (const float*)d_in[16];
    const float* s4_logdt = (const float*)d_in[17];
    const float* s4_Wo    = (const float*)d_in[18];
    const float* s4_bo    = (const float*)d_in[19];
    const float* res_ln_g = (const float*)d_in[20];
    const float* res_ln_b = (const float*)d_in[21];
    const float* res_W1   = (const float*)d_in[22];
    const float* res_b1   = (const float*)d_in[23];
    const float* res_W2   = (const float*)d_in[24];
    const float* res_b2   = (const float*)d_in[25];
    const float* W_out    = (const float*)d_in[26];
    const float* b_out    = (const float*)d_in[27];

    // workspace layout
    char* ws = (char*)d_ws;
    const size_t H_BYTES = (size_t)M_ * H_ * 4;        // 131,072,000  f32 h
    const size_t A_BYTES = (size_t)M_ * H_ * 2;        //  65,536,000  bf16 bufA
    const size_t B_BYTES = (size_t)M_ * 2 * H_ * 2;    // 131,072,000  bf16 bufB
    const size_t K_BYTES = (size_t)4 * H_ * SP_ * 2;   //     524,288  bf16 kernels
    float*  h    = (float*) (ws);
    __bf16* bufA = (__bf16*)(ws + H_BYTES);
    __bf16* bufB = (__bf16*)(ws + H_BYTES + A_BYTES);          // also zt [B][H][SP_]
    __bf16* Kker = (__bf16*)(ws + H_BYTES + A_BYTES + B_BYTES);
    float*  te   = (float*) (ws + H_BYTES + A_BYTES + B_BYTES + K_BYTES);

    // 1) synthesize S4 conv kernels (all 4 layers)
    k_s4_kernel<<<4 * H_, 128, 0, stream>>>(s4_logA, s4_Aim, s4_Cre, s4_Cim,
                                            s4_logdt, Kker);
    // 2) input embedding
    k_embed<<<M_, 128, 0, stream>>>(x, mask, observed, W_in, b_in, h);

    // 3) S4D layers
    for (int l = 0; l < 4; ++l) {
        k_ln<<<M_, 128, 0, stream>>>(h, s4_ln_g + l * H_, s4_ln_b + l * H_,
                                     bufB, /*mode=*/1);
        k_conv<<<dim3(B_ / 64, SP_ / 16, H_), 32, 0, stream>>>(
            bufB, Kker + (size_t)l * H_ * SP_, s4_D + l * H_, bufA);
        k_gemm<<<dim3(M_ / 128, 1), 256, 128 * (128 + 8) * 2, stream>>>(
            bufA, 128, s4_Wo + (size_t)l * H_ * H_, 128, s4_bo + l * H_,
            h, nullptr, /*gelu=*/0);
    }

    // 4) diffusion time embedding
    k_time_mlp<<<B_, 256, 0, stream>>>(tsteps, time_W1, time_b1, time_W2,
                                       time_b2, te);
    k_add_te<<<(M_ * H_) / 256, 256, 0, stream>>>(h, te);

    // 5) residual MLP blocks
    for (int l = 0; l < 6; ++l) {
        k_ln<<<M_, 128, 0, stream>>>(h, res_ln_g + l * H_, res_ln_b + l * H_,
                                     bufA, /*mode=*/0);
        k_gemm<<<dim3(M_ / 128, 2), 256, 128 * (128 + 8) * 2, stream>>>(
            bufA, 128, res_W1 + (size_t)l * H_ * 2 * H_, 256, res_b1 + l * 2 * H_,
            nullptr, bufB, /*gelu=*/1);
        k_gemm<<<dim3(M_ / 128, 1), 256, 128 * (256 + 8) * 2, stream>>>(
            bufB, 256, res_W2 + (size_t)l * 2 * H_ * H_, 128, res_b2 + l * H_,
            h, nullptr, /*gelu=*/0);
    }

    // 6) output projection
    k_out<<<M_ / 16, 224, 0, stream>>>(h, W_out, b_out, (float*)d_out);
}
